// SimpleLanguageModel_61125974557024
// MI455X (gfx1250) — compile-verified, matrix operand order checked by reading, not run on hardware
//
#include <hip/hip_runtime.h>
#include <hip/hip_bf16.h>
#include <stdint.h>

#define HIDDEN 512
#define G3     1536     // 3*HIDDEN
#define BATCH  8
#define TLEN   512
#define VOCABN 32000
#define MROWS  4096     // BATCH*TLEN

#define KDIM   512
#define KC     32       // K-chunk staged in LDS
#define NCH    (KDIM / KC)
#define LDW    40       // padded LDS row stride (ushorts): 32 data + 8 pad -> conflict-free

typedef __attribute__((ext_vector_type(16))) __bf16 v16bf;
typedef __attribute__((ext_vector_type(8)))  float  v8f;

union BF16x16 { v16bf v; uint32_t u[8]; };

#if defined(__has_builtin)
#if __has_builtin(__builtin_amdgcn_global_load_async_to_lds_b128)
#define HAVE_ASYNC_LDS 1
#endif
#endif
#ifndef HAVE_ASYNC_LDS
#define HAVE_ASYNC_LDS 0
#endif

__device__ __forceinline__ unsigned short f32_to_bf16(float f) {
    union { float f; uint32_t u; } c; c.f = f;
    uint32_t u = c.u;
    u = (u + 0x7FFFu + ((u >> 16) & 1u)) >> 16;   // round-to-nearest-even
    return (unsigned short)u;
}

__device__ __forceinline__ uint32_t f32x2_to_bf16x2(float lo, float hi) {
    return ((uint32_t)f32_to_bf16(hi) << 16) | (uint32_t)f32_to_bf16(lo);
}

// K offset for dword j of a 16x32 bf16 A/B fragment (ISA 7.12.2):
// VGPR0: K=0,1 (lanes0-15) / K=8,9 (lanes16-31); VGPR4: K=16,17 / 24,25; etc.
__device__ __forceinline__ int frag_k(int kbase, int half, int j) {
    return kbase + (half << 3) + ((j < 4) ? (j << 1) : (16 + ((j - 4) << 1)));
}

#if HAVE_ASYNC_LDS
typedef __attribute__((__vector_size__(4 * sizeof(int)))) int v4i_;
typedef __attribute__((address_space(1))) v4i_ gv4i;
typedef __attribute__((address_space(3))) v4i_ lv4i;

__device__ __forceinline__ void async_cp16(const void* g, void* l) {
    __builtin_amdgcn_global_load_async_to_lds_b128((gv4i*)g, (lv4i*)l, 0, 0);
}
template<int N>
__device__ __forceinline__ void wait_asynccnt() {
#if __has_builtin(__builtin_amdgcn_s_wait_asynccnt)
    __builtin_amdgcn_s_wait_asynccnt(N);
#else
    asm volatile("s_wait_asynccnt %0" :: "i"(N) : "memory");
#endif
}
#endif

// ---------------------------------------------------------------- converts
__global__ void k_f32_to_bf16(const float* __restrict__ src,
                              uint32_t* __restrict__ dst, int npairs) {
    int i = blockIdx.x * blockDim.x + threadIdx.x;
    int stride = gridDim.x * blockDim.x;
    for (; i < npairs; i += stride) {
        float2 f = ((const float2*)src)[i];
        dst[i] = f32x2_to_bf16x2(f.x, f.y);
    }
}

// embedding lookup -> bf16 activations, row m = b*T + t
__global__ void k_embed(const int* __restrict__ ids,
                        const float* __restrict__ emb,
                        uint32_t* __restrict__ xbf) {
    const int total = MROWS * (HIDDEN / 2);
    int i = blockIdx.x * blockDim.x + threadIdx.x;
    int stride = gridDim.x * blockDim.x;
    for (; i < total; i += stride) {
        int row = i / (HIDDEN / 2);
        int kp  = i % (HIDDEN / 2);
        int id  = ids[row];
        float2 f = ((const float2*)(emb + (size_t)id * HIDDEN))[kp];
        xbf[i] = f32x2_to_bf16x2(f.x, f.y);
    }
}

// ------------------------------------------------------ LDS-staged WMMA GEMM
// out[m,n] = sum_k A[m,k]*W[n,k] + bias[n]   (A: MxK bf16, W: NxK bf16, K=512)
// Block: 256 threads, tile 128x128. 8 waves in 2(M)x4(N); wave tile 64x32
// (4 M-tiles x 2 N-tiles). A/B K-chunks (128x32 bf16) double-buffered in LDS.
__global__ __launch_bounds__(256) void k_gemm_bf16(
    const unsigned short* __restrict__ A, const unsigned short* __restrict__ W,
    const float* __restrict__ bias, float* __restrict__ out, int N) {

    __shared__ unsigned short sA[2][128 * LDW];   // 2 x 10 KB
    __shared__ unsigned short sB[2][128 * LDW];   // 2 x 10 KB

    const int tid  = threadIdx.x;
    const int lane = tid & 31;
    const int wave = tid >> 5;
    const int half = lane >> 4;
    const int lq   = lane & 15;
    const int wm   = wave >> 2;          // 0..1
    const int wn   = wave & 3;           // 0..3
    const int blockM = blockIdx.x * 128;
    const int blockN = blockIdx.y * 128;

    // copy mapping: thread -> (row cr, 32B half cq) of the 128x32 chunk
    const int cr = tid & 127;
    const int cq = tid >> 7;             // 0..1
    const unsigned short* gA = A + (size_t)(blockM + cr) * KDIM + cq * 16;
    const unsigned short* gW = W + (size_t)(blockN + cr) * KDIM + cq * 16;
    const int lofs = cr * LDW + cq * 16;

    v8f acc[8] = {};

#if HAVE_ASYNC_LDS
    // ---- async double-buffered pipeline (4 async b128 per thread per chunk)
    async_cp16(gA,     &sA[0][lofs]);
    async_cp16(gA + 8, &sA[0][lofs + 8]);
    async_cp16(gW,     &sB[0][lofs]);
    async_cp16(gW + 8, &sB[0][lofs + 8]);
    for (int c = 0; c < NCH; ++c) {
        if (c + 1 < NCH) {
            const int kb = (c + 1) * KC;
            const int nb = (c + 1) & 1;
            async_cp16(gA + kb,     &sA[nb][lofs]);
            async_cp16(gA + kb + 8, &sA[nb][lofs + 8]);
            async_cp16(gW + kb,     &sB[nb][lofs]);
            async_cp16(gW + kb + 8, &sB[nb][lofs + 8]);
            wait_asynccnt<4>();       // current chunk's 4 copies done (in-order)
        } else {
            wait_asynccnt<0>();
        }
        __syncthreads();
        {
            const int buf = c & 1;
            const unsigned short* bAp = sA[buf];
            const unsigned short* bBp = sB[buf];
            BF16x16 af[4], bfr[2];
            #pragma unroll
            for (int mt = 0; mt < 4; ++mt) {
                const unsigned short* rp = bAp + (wm * 64 + mt * 16 + lq) * LDW + half * 8;
                *(uint4*)&af[mt].u[0] = *(const uint4*)(rp);
                *(uint4*)&af[mt].u[4] = *(const uint4*)(rp + 16);
            }
            #pragma unroll
            for (int nt = 0; nt < 2; ++nt) {
                const unsigned short* rp = bBp + (wn * 32 + nt * 16 + lq) * LDW + half * 8;
                *(uint4*)&bfr[nt].u[0] = *(const uint4*)(rp);
                *(uint4*)&bfr[nt].u[4] = *(const uint4*)(rp + 16);
            }
            #pragma unroll
            for (int mt = 0; mt < 4; ++mt)
                #pragma unroll
                for (int nt = 0; nt < 2; ++nt)
                    acc[mt * 2 + nt] = __builtin_amdgcn_wmma_f32_16x16x32_bf16(
                        false, af[mt].v, false, bfr[nt].v, (short)0,
                        acc[mt * 2 + nt], false, false);
        }
        __syncthreads();
    }
#else
    // ---- register-staged pipeline (fallback)
    uint4 ra0, ra1, rb0, rb1;
    ra0 = *(const uint4*)(gA);
    ra1 = *(const uint4*)(gA + 8);
    rb0 = *(const uint4*)(gW);
    rb1 = *(const uint4*)(gW + 8);
    for (int c = 0; c < NCH; ++c) {
        const int buf = c & 1;
        *(uint4*)&sA[buf][lofs]     = ra0;
        *(uint4*)&sA[buf][lofs + 8] = ra1;
        *(uint4*)&sB[buf][lofs]     = rb0;
        *(uint4*)&sB[buf][lofs + 8] = rb1;
        if (c + 1 < NCH) {
            const int kb = (c + 1) * KC;
            ra0 = *(const uint4*)(gA + kb);
            ra1 = *(const uint4*)(gA + kb + 8);
            rb0 = *(const uint4*)(gW + kb);
            rb1 = *(const uint4*)(gW + kb + 8);
        }
        __syncthreads();
        {
            const unsigned short* bAp = sA[buf];
            const unsigned short* bBp = sB[buf];
            BF16x16 af[4], bfr[2];
            #pragma unroll
            for (int mt = 0; mt < 4; ++mt) {
                const unsigned short* rp = bAp + (wm * 64 + mt * 16 + lq) * LDW + half * 8;
                *(uint4*)&af[mt].u[0] = *(const uint4*)(rp);
                *(uint4*)&af[mt].u[4] = *(const uint4*)(rp + 16);
            }
            #pragma unroll
            for (int nt = 0; nt < 2; ++nt) {
                const unsigned short* rp = bBp + (wn * 32 + nt * 16 + lq) * LDW + half * 8;
                *(uint4*)&bfr[nt].u[0] = *(const uint4*)(rp);
                *(uint4*)&bfr[nt].u[4] = *(const uint4*)(rp + 16);
            }
            #pragma unroll
            for (int mt = 0; mt < 4; ++mt)
                #pragma unroll
                for (int nt = 0; nt < 2; ++nt)
                    acc[mt * 2 + nt] = __builtin_amdgcn_wmma_f32_16x16x32_bf16(
                        false, af[mt].v, false, bfr[nt].v, (short)0,
                        acc[mt * 2 + nt], false, false);
        }
        __syncthreads();
    }
#endif

    // ---- epilogue: bias + store
    #pragma unroll
    for (int mt = 0; mt < 4; ++mt) {
        #pragma unroll
        for (int nt = 0; nt < 2; ++nt) {
            const int col = blockN + wn * 32 + nt * 16 + lq;
            const float bv = bias[col];
            const v8f a = acc[mt * 2 + nt];
            #pragma unroll
            for (int r = 0; r < 8; ++r) {
                const int m = blockM + wm * 64 + mt * 16 + r + half * 8;
                out[(size_t)m * N + col] = a[r] + bv;
            }
        }
    }
}

// ------------------------------------------- persistent GRU scan + LayerNorm
__global__ __launch_bounds__(1024) void k_gru_ln(
    const float* __restrict__ xg, const unsigned short* __restrict__ whh,
    const float* __restrict__ bhh, const float* __restrict__ gamma,
    const float* __restrict__ beta, unsigned short* __restrict__ ybf) {
    __shared__ float hg[BATCH][G3];                 // 48 KB
    __shared__ unsigned short hbf[BATCH][HIDDEN];   // 8 KB
    __shared__ float red1[32], red2[32];
    __shared__ float mu_s[BATCH], rs_s[BATCH];

    const int tid  = threadIdx.x;
    const int lane = tid & 31;
    const int wave = tid >> 5;            // 0..31
    const int half = lane >> 4;
    const int lq   = lane & 15;
    const int brow = tid >> 7;            // batch row 0..7
    const int j0   = (tid & 127) * 4;     // 4 owned hidden elements

    float h[4] = {0.f, 0.f, 0.f, 0.f};
    for (int i = tid; i < BATCH * HIDDEN; i += 1024)
        ((unsigned short*)hbf)[i] = 0;
    __syncthreads();

    for (int t = 0; t < TLEN; ++t) {
        // ---- recurrent GEMM (rows 8..15 of A are zero padding)
        v8f acc[3] = {};
        for (int kb = 0; kb < 16; ++kb) {
            const int kbase = kb * 32;
            BF16x16 a;
            if (lq < 8) {
                #pragma unroll
                for (int j = 0; j < 8; ++j)
                    a.u[j] = *(const uint32_t*)&hbf[lq][frag_k(kbase, half, j)];
            } else {
                #pragma unroll
                for (int j = 0; j < 8; ++j) a.u[j] = 0u;
            }
            #pragma unroll
            for (int i = 0; i < 3; ++i) {
                const int gb = (wave * 3 + i) * 16;
                const unsigned short* Wr = whh + (size_t)(gb + lq) * HIDDEN;
                BF16x16 b;
                #pragma unroll
                for (int j = 0; j < 8; ++j)
                    b.u[j] = *(const uint32_t*)(Wr + frag_k(kbase, half, j));
                acc[i] = __builtin_amdgcn_wmma_f32_16x16x32_bf16(
                    false, a.v, false, b.v, (short)0, acc[i], false, false);
            }
        }
        if (half == 0) {
            #pragma unroll
            for (int i = 0; i < 3; ++i) {
                const int gb = (wave * 3 + i) * 16;
                #pragma unroll
                for (int r = 0; r < 8; ++r) hg[r][gb + lq] = acc[i][r];
            }
        }
        __syncthreads();

        // ---- gates + hidden update (gate order r,z,n)
        const float* xrow = xg + ((size_t)brow * TLEN + t) * G3;
        float s1 = 0.f, s2 = 0.f;
        #pragma unroll
        for (int e = 0; e < 4; ++e) {
            const int hx = j0 + e;
            const float xr = xrow[hx];
            const float xz = xrow[HIDDEN + hx];
            const float xn = xrow[2 * HIDDEN + hx];
            const float hr = hg[brow][hx]              + bhh[hx];
            const float hz = hg[brow][HIDDEN + hx]     + bhh[HIDDEN + hx];
            const float hn = hg[brow][2 * HIDDEN + hx] + bhh[2 * HIDDEN + hx];
            const float r = 1.f / (1.f + __expf(-(xr + hr)));
            const float z = 1.f / (1.f + __expf(-(xz + hz)));
            const float n = tanhf(xn + r * hn);
            h[e] = (1.f - z) * n + z * h[e];
            hbf[brow][hx] = f32_to_bf16(h[e]);
            s1 += h[e];
            s2 += h[e] * h[e];
        }
        // ---- LayerNorm reductions: wave-level shuffle, then 4 partials/row
        #pragma unroll
        for (int off = 16; off > 0; off >>= 1) {
            s1 += __shfl_down(s1, off, 32);
            s2 += __shfl_down(s2, off, 32);
        }
        if (lane == 0) { red1[wave] = s1; red2[wave] = s2; }
        __syncthreads();
        if (tid < 8) {
            const float a1 = red1[tid*4] + red1[tid*4+1] + red1[tid*4+2] + red1[tid*4+3];
            const float a2 = red2[tid*4] + red2[tid*4+1] + red2[tid*4+2] + red2[tid*4+3];
            const float mu  = a1 * (1.0f / HIDDEN);
            const float var = a2 * (1.0f / HIDDEN) - mu * mu;
            mu_s[tid] = mu;
            rs_s[tid] = rsqrtf(var + 1e-5f);
        }
        __syncthreads();
        {
            const float mu = mu_s[brow], rs = rs_s[brow];
            uint32_t* yrow = (uint32_t*)(ybf + ((size_t)brow * TLEN + t) * HIDDEN);
            #pragma unroll
            for (int e = 0; e < 4; e += 2) {
                const int hx = j0 + e;
                const float y0 = (h[e]   - mu) * rs * gamma[hx]   + beta[hx];
                const float y1 = (h[e+1] - mu) * rs * gamma[hx+1] + beta[hx+1];
                yrow[hx >> 1] = f32x2_to_bf16x2(y0, y1);
            }
        }
        __syncthreads();
    }
}

// ---------------------------------------------------------------- launcher
extern "C" void kernel_launch(void* const* d_in, const int* in_sizes, int n_in,
                              void* d_out, int out_size, void* d_ws, size_t ws_size,
                              hipStream_t stream) {
    (void)in_sizes; (void)n_in; (void)out_size; (void)ws_size;
    const int*   ids   = (const int*)  d_in[0];
    const float* emb   = (const float*)d_in[1];
    const float* w_ih  = (const float*)d_in[2];
    const float* w_hh  = (const float*)d_in[3];
    const float* b_ih  = (const float*)d_in[4];
    const float* b_hh  = (const float*)d_in[5];
    const float* gamma = (const float*)d_in[6];
    const float* beta  = (const float*)d_in[7];
    const float* headw = (const float*)d_in[8];
    const float* headb = (const float*)d_in[9];
    float* out = (float*)d_out;

    char* ws = (char*)d_ws;
    size_t off = 0;
    auto alloc = [&](size_t bytes) -> void* {
        void* p = ws + off;
        off = (off + bytes + 255) & ~(size_t)255;
        return p;
    };
    unsigned short* xbf   = (unsigned short*)alloc((size_t)MROWS * HIDDEN * 2);
    unsigned short* wihbf = (unsigned short*)alloc((size_t)G3 * HIDDEN * 2);
    unsigned short* whhbf = (unsigned short*)alloc((size_t)G3 * HIDDEN * 2);
    unsigned short* hwbf  = (unsigned short*)alloc((size_t)VOCABN * HIDDEN * 2);
    float*          xgbuf = (float*)alloc((size_t)MROWS * G3 * 4);
    unsigned short* ybf   = (unsigned short*)alloc((size_t)MROWS * HIDDEN * 2);

    const int npw = G3 * HIDDEN / 2;
    k_f32_to_bf16<<<(npw + 255) / 256, 256, 0, stream>>>(w_ih, (uint32_t*)wihbf, npw);
    k_f32_to_bf16<<<(npw + 255) / 256, 256, 0, stream>>>(w_hh, (uint32_t*)whhbf, npw);
    const int nph = VOCABN * HIDDEN / 2;
    k_f32_to_bf16<<<8192, 256, 0, stream>>>(headw, (uint32_t*)hwbf, nph);

    k_embed<<<(MROWS * HIDDEN / 2 + 255) / 256, 256, 0, stream>>>(ids, emb, (uint32_t*)xbf);

    dim3 g1(MROWS / 128, G3 / 128);       // (32, 12)
    k_gemm_bf16<<<g1, 256, 0, stream>>>(xbf, wihbf, b_ih, xgbuf, G3);

    k_gru_ln<<<1, 1024, 0, stream>>>(xgbuf, whhbf, b_hh, gamma, beta, ybf);

    dim3 g2(MROWS / 128, VOCABN / 128);   // (32, 250)
    k_gemm_bf16<<<g2, 256, 0, stream>>>(ybf, hwbf, headb, out, VOCABN);
}
